// CoverageAttention_61641370632426
// MI455X (gfx1250) — matrix operand order repaired and checked by reading, not compiled, not run on hardware
//
#include <hip/hip_runtime.h>
#include <hip/hip_bf16.h>

#define B_    64
#define S_    2048
#define H_    1024
#define E_    2048          // = 2H, GEMM K dimension
#define KT_   (E_ / 32)     // 64 k-tiles of 32
#define ROWS_ 64            // M rows per workgroup in the scores GEMM

typedef __attribute__((ext_vector_type(16))) __bf16 v16bf;
typedef __attribute__((ext_vector_type(8)))  float  v8f;

__device__ __forceinline__ unsigned pack_bf16_rne(float lo, float hi) {
  unsigned a = __builtin_bit_cast(unsigned, lo);
  unsigned b = __builtin_bit_cast(unsigned, hi);
  a = (a + 0x7FFFu + ((a >> 16) & 1u)) >> 16;
  b = (b + 0x7FFFu + ((b >> 16) & 1u)) >> 16;
  return a | (b << 16);
}

// ---------------------------------------------------------------------------
// K0: repack W_h (f32 row-major [E_, H_]) into bf16 WMMA-B tile layout:
//     dst[(kt*H_ + n)*32 + k]  (bf16), k = 0..31 contiguous per (kt, n).
// ---------------------------------------------------------------------------
__global__ __launch_bounds__(256) void pack_wh_kernel(
    const float* __restrict__ Wh, unsigned* __restrict__ dst) {
  int gid = blockIdx.x * 256 + threadIdx.x;   // 0 .. KT_*H_-1
  int kt = gid >> 10;
  int n  = gid & (H_ - 1);
  unsigned out[16];
#pragma unroll
  for (int p = 0; p < 16; ++p) {
    float lo = Wh[(size_t)(kt * 32 + 2 * p    ) * H_ + n];
    float hi = Wh[(size_t)(kt * 32 + 2 * p + 1) * H_ + n];
    out[p] = pack_bf16_rne(lo, hi);
  }
  uint4* d4 = (uint4*)(dst + (size_t)gid * 16);
  d4[0] = make_uint4(out[0],  out[1],  out[2],  out[3]);
  d4[1] = make_uint4(out[4],  out[5],  out[6],  out[7]);
  d4[2] = make_uint4(out[8],  out[9],  out[10], out[11]);
  d4[3] = make_uint4(out[12], out[13], out[14], out[15]);
}

// ---------------------------------------------------------------------------
// K1: dec_proj[b, :] = decoder_hidden[b, :] @ W_d   (tiny; VALU is fine)
// ---------------------------------------------------------------------------
__global__ __launch_bounds__(256) void decproj_kernel(
    const float* __restrict__ dh, const float* __restrict__ Wd,
    float* __restrict__ decp) {
  __shared__ float sdh[H_];
  const int b = blockIdx.x, t = threadIdx.x;
  for (int i = t; i < H_; i += 256) sdh[i] = dh[b * H_ + i];
  __syncthreads();
  float acc[4] = {0.f, 0.f, 0.f, 0.f};
  for (int k = 0; k < H_; ++k) {
    float d = sdh[k];
#pragma unroll
    for (int j = 0; j < 4; ++j)
      acc[j] += d * Wd[(size_t)k * H_ + t + j * 256];
  }
#pragma unroll
  for (int j = 0; j < 4; ++j) decp[b * H_ + t + j * 256] = acc[j];
}

// ---------------------------------------------------------------------------
// K2: scores GEMM, M=64 x N=256 per workgroup (512 threads, 16 waves).
//   Each wave: ONE 16-col B fragment per k-step, reused across FOUR A
//   row-tiles (4 WMMAs) -> 4x less L2 traffic for W_h than M=16 tiling.
//   A strip (64x32) staged in LDS (double buffered), branch-free body.
//   Each WG emits a partial score plane for its N-chunk (summed in K3).
// ---------------------------------------------------------------------------
__global__ __launch_bounds__(512) void scores_kernel(
    const float* __restrict__ enc, const float* __restrict__ cov,
    const unsigned* __restrict__ whp, const float* __restrict__ decp,
    const float* __restrict__ Wc, const float* __restrict__ vvec,
    float* __restrict__ scorPart) {
  __shared__ __align__(16) unsigned short sAh[2][ROWS_ * 32];  // 2 x 4 KB
  __shared__ float sCov[ROWS_];
  __shared__ float sScore[ROWS_];

  const int tid   = threadIdx.x;
  const int lane  = tid & 31;
  const int wave  = tid >> 5;       // 0..15
  const int nhalf = lane >> 4;      // 0: lanes 0-15, 1: lanes 16-31
  const int nl    = lane & 15;

  const int blk    = blockIdx.x;
  const int nchunk = blk & 3;                 // which 256 columns
  const int strip  = blk >> 2;                // row strip (64 rows)
  const int b      = strip >> 5;              // S_/ROWS_ = 32 strips per b
  const int s0     = (strip & 31) * ROWS_;

  if (tid < ROWS_) {
    sCov[tid]   = cov[b * S_ + s0 + tid];
    sScore[tid] = 0.f;
  }

  v8f acc[4] = {};   // 4 row-tiles x 1 col-tile per wave

  // cooperative A staging: thread owns (row = tid>>3, 4 k's at (tid&7)*4)
  const int am = tid >> 3;
  const int ak = (tid & 7) * 4;
  const float* aptr = enc + ((size_t)b * S_ + s0 + am) * E_ + ak;
  const int sbase = am * 32 + ak;

  // the single column this lane feeds (per 16-col tile)
  const int ncol = (nchunk << 8) + (wave << 4) + nl;
  const int boff = nhalf;

  // prologue: stage tile kt=0
  {
    float4 f = *(const float4*)aptr;
    *(uint2*)&sAh[0][sbase] =
        make_uint2(pack_bf16_rne(f.x, f.y), pack_bf16_rne(f.z, f.w));
  }
  __syncthreads();

  int pb = 0;
  for (int kt = 0; kt < KT_ - 1; ++kt) {
    float4 fnext = *(const float4*)(aptr + (size_t)(kt + 1) * 32);

    // one B fragment per wave (16 cols, K=0..31 contiguous per column)
    const uint4* bp = (const uint4*)(whp + ((size_t)(kt << 10) + ncol) * 16);
    union { uint4 u[2]; v16bf v; } ub;
    ub.u[0] = bp[boff];
    ub.u[1] = bp[boff + 2];

    // four A fragments (row-tiles mi*16), 16-bit A 16x32 layout:
    //   lanes 0-15: K0-7 / K16-23 ; lanes 16-31: K8-15 / K24-31
    const uint4* ap = (const uint4*)(&sAh[pb][0]);  // row stride = 4 uint4
    union { uint4 u[2]; v16bf v; } ua0, ua1, ua2, ua3;
    {
      const int r0 = (nl) * 4 + nhalf;
      ua0.u[0] = ap[r0];           ua0.u[1] = ap[r0 + 2];
      ua1.u[0] = ap[r0 + 64];      ua1.u[1] = ap[r0 + 66];
      ua2.u[0] = ap[r0 + 128];     ua2.u[1] = ap[r0 + 130];
      ua3.u[0] = ap[r0 + 192];     ua3.u[1] = ap[r0 + 194];
    }

    acc[0] = __builtin_amdgcn_wmma_f32_16x16x32_bf16(
        false, ua0.v, false, ub.v, (short)0, acc[0], false, false);
    acc[1] = __builtin_amdgcn_wmma_f32_16x16x32_bf16(
        false, ua1.v, false, ub.v, (short)0, acc[1], false, false);
    acc[2] = __builtin_amdgcn_wmma_f32_16x16x32_bf16(
        false, ua2.v, false, ub.v, (short)0, acc[2], false, false);
    acc[3] = __builtin_amdgcn_wmma_f32_16x16x32_bf16(
        false, ua3.v, false, ub.v, (short)0, acc[3], false, false);

    *(uint2*)&sAh[pb ^ 1][sbase] =
        make_uint2(pack_bf16_rne(fnext.x, fnext.y),
                   pack_bf16_rne(fnext.z, fnext.w));
    __syncthreads();
    pb ^= 1;
  }

  // peeled last k-tile
  {
    const int kt = KT_ - 1;
    const uint4* bp = (const uint4*)(whp + ((size_t)(kt << 10) + ncol) * 16);
    union { uint4 u[2]; v16bf v; } ub;
    ub.u[0] = bp[boff];
    ub.u[1] = bp[boff + 2];

    const uint4* ap = (const uint4*)(&sAh[pb][0]);
    union { uint4 u[2]; v16bf v; } ua0, ua1, ua2, ua3;
    {
      const int r0 = (nl) * 4 + nhalf;
      ua0.u[0] = ap[r0];           ua0.u[1] = ap[r0 + 2];
      ua1.u[0] = ap[r0 + 64];      ua1.u[1] = ap[r0 + 66];
      ua2.u[0] = ap[r0 + 128];     ua2.u[1] = ap[r0 + 130];
      ua3.u[0] = ap[r0 + 192];     ua3.u[1] = ap[r0 + 194];
    }

    acc[0] = __builtin_amdgcn_wmma_f32_16x16x32_bf16(
        false, ua0.v, false, ub.v, (short)0, acc[0], false, false);
    acc[1] = __builtin_amdgcn_wmma_f32_16x16x32_bf16(
        false, ua1.v, false, ub.v, (short)0, acc[1], false, false);
    acc[2] = __builtin_amdgcn_wmma_f32_16x16x32_bf16(
        false, ua2.v, false, ub.v, (short)0, acc[2], false, false);
    acc[3] = __builtin_amdgcn_wmma_f32_16x16x32_bf16(
        false, ua3.v, false, ub.v, (short)0, acc[3], false, false);
  }

  // epilogue: + dec_proj + cov*Wc, tanh, partial dot with v over this column
  const float dpn = decp[b * H_ + ncol];
  const float wcn = Wc[ncol];
  const float vn  = vvec[ncol];
#pragma unroll
  for (int mi = 0; mi < 4; ++mi) {
    float scr[8];
#pragma unroll
    for (int r = 0; r < 8; ++r) {
      const int m = mi * 16 + r + (nhalf << 3);   // row within strip
      float e = acc[mi][r] + dpn + sCov[m] * wcn;
      e = tanhf(e);
      scr[r] = e * vn;
    }
    // sum the 16 columns of this wave's tile
#pragma unroll
    for (int r = 0; r < 8; ++r) {
      float x = scr[r];
      x += __shfl_xor(x, 1);
      x += __shfl_xor(x, 2);
      x += __shfl_xor(x, 4);
      x += __shfl_xor(x, 8);
      scr[r] = x;
    }
    if (nl == 0) {
#pragma unroll
      for (int r = 0; r < 8; ++r)
        atomicAdd(&sScore[mi * 16 + r + (nhalf << 3)], scr[r]);  // ds_add_f32
    }
  }
  __syncthreads();
  if (tid < ROWS_) {
    // partial plane for this N-chunk; summed (and masked) in softmax kernel
    scorPart[(size_t)nchunk * (B_ * S_) + b * S_ + s0 + tid] = sScore[tid];
  }
}

// ---------------------------------------------------------------------------
// K3: sum 4 partial score planes, mask, softmax over S per batch;
//     writes attn_weights and coverage_new into d_out.
// ---------------------------------------------------------------------------
__global__ __launch_bounds__(256) void softmax_kernel(
    const float* __restrict__ scorPart, const int* __restrict__ maskp,
    const float* __restrict__ cov, float* __restrict__ out) {
  __shared__ float red[256];
  const int b = blockIdx.x, t = threadIdx.x;
  float vals[8];
  float lmax = -3.4e38f;
#pragma unroll
  for (int i = 0; i < 8; ++i) {
    const int s = t + i * 256;
    float x = scorPart[(size_t)0 * (B_ * S_) + b * S_ + s]
            + scorPart[(size_t)1 * (B_ * S_) + b * S_ + s]
            + scorPart[(size_t)2 * (B_ * S_) + b * S_ + s]
            + scorPart[(size_t)3 * (B_ * S_) + b * S_ + s];
    if (maskp[b * S_ + s] == 0) x = -10000.0f;
    vals[i] = x;
    lmax = fmaxf(lmax, x);
  }
  red[t] = lmax;
  __syncthreads();
  for (int o = 128; o > 0; o >>= 1) {
    if (t < o) red[t] = fmaxf(red[t], red[t + o]);
    __syncthreads();
  }
  const float m = red[0];
  __syncthreads();
  float lsum = 0.f;
#pragma unroll
  for (int i = 0; i < 8; ++i) {
    vals[i] = __expf(vals[i] - m);
    lsum += vals[i];
  }
  red[t] = lsum;
  __syncthreads();
  for (int o = 128; o > 0; o >>= 1) {
    if (t < o) red[t] += red[t + o];
    __syncthreads();
  }
  const float inv = 1.0f / red[0];
  float* attn = out + (size_t)B_ * E_ + b * S_;
  float* covn = out + (size_t)B_ * E_ + (size_t)B_ * S_ + b * S_;
  const float* cv = cov + b * S_;
#pragma unroll
  for (int i = 0; i < 8; ++i) {
    const float w = vals[i] * inv;
    attn[t + i * 256] = w;
    covn[t + i * 256] = cv[t + i * 256] + w;
  }
}

// ---------------------------------------------------------------------------
// K4: context[b, e] = sum_s attn[b, s] * enc[b, s, e]   (memory bound)
// ---------------------------------------------------------------------------
__global__ __launch_bounds__(256) void context_kernel(
    const float* __restrict__ enc, const float* __restrict__ attn_all,
    float* __restrict__ ctx) {
  __shared__ float sa[256];
  const int b  = blockIdx.x >> 3;
  const int ec = blockIdx.x & 7;
  const int t  = threadIdx.x;
  const int e  = ec * 256 + t;
  const float* attn = attn_all + b * S_;
  float acc = 0.f;
  for (int s0 = 0; s0 < S_; s0 += 256) {
    __syncthreads();
    sa[t] = attn[s0 + t];
    __syncthreads();
    const float* ep = enc + ((size_t)b * S_ + s0) * E_ + e;
#pragma unroll 8
    for (int i = 0; i < 256; ++i) acc += sa[i] * ep[(size_t)i * E_];
  }
  ctx[(size_t)b * E_ + e] = acc;
}

// ---------------------------------------------------------------------------
extern "C" void kernel_launch(void* const* d_in, const int* in_sizes, int n_in,
                              void* d_out, int out_size, void* d_ws, size_t ws_size,
                              hipStream_t stream) {
  (void)in_sizes; (void)n_in; (void)out_size; (void)ws_size;
  const float* dec_hidden = (const float*)d_in[0];   // (B, H)
  const float* enc        = (const float*)d_in[1];   // (B, S, 2H)
  const float* cov        = (const float*)d_in[2];   // (B, S)
  const int*   mask       = (const int*)  d_in[3];   // (B, S)
  const float* Wh         = (const float*)d_in[4];   // (2H, H)
  const float* Wd         = (const float*)d_in[5];   // (H, H)
  const float* Wc         = (const float*)d_in[6];   // (1, H)
  const float* v          = (const float*)d_in[7];   // (H, 1)
  float* out = (float*)d_out;   // [context (B,2H) | attn (B,S) | cov_new (B,S)]

  unsigned char* ws = (unsigned char*)d_ws;
  unsigned* whp  = (unsigned*)ws;                                   // 4 MB packed W_h
  float*    decp = (float*)(ws + (size_t)E_ * H_ * 2);              // 256 KB
  float*    scor = (float*)(ws + (size_t)E_ * H_ * 2 + (size_t)B_ * H_ * 4); // 4 planes, 2 MB

  pack_wh_kernel <<<(KT_ * H_) / 256, 256, 0, stream>>>(Wh, whp);
  decproj_kernel <<<B_, 256, 0, stream>>>(dec_hidden, Wd, decp);
  scores_kernel  <<<(B_ * S_ / ROWS_) * 4, 512, 0, stream>>>(
      enc, cov, whp, decp, Wc, v, scor);
  softmax_kernel <<<B_, 256, 0, stream>>>(scor, mask, cov, out);
  context_kernel <<<B_ * (E_ / 256), 256, 0, stream>>>(
      enc, out + (size_t)B_ * E_, out);
}
